// LlamaAttention_7662221656480
// MI455X (gfx1250) — matrix operand order repaired and checked by reading, not compiled, not run on hardware
//
#include <hip/hip_runtime.h>

// ---- problem constants (match reference) ----
#define BB    2
#define SS    2048
#define HID   2048
#define NH    32
#define NKVH  8
#define HD    64
#define KVDIM (NKVH * HD)   // 512

typedef __attribute__((ext_vector_type(16))) _Float16 v16h;
typedef __attribute__((ext_vector_type(8)))  float    v8f;

union Frag16 { v16h h; unsigned u[8]; };

__device__ __forceinline__ v8f wmma_f16(const Frag16& a, const Frag16& b, v8f c) {
  return __builtin_amdgcn_wmma_f32_16x16x32_f16(false, a.h, false, b.h, (short)0, c,
                                                false, false);
}

// A-fragment (16x32, MxK): lane lo = M row (caller adds lo*stride), lh = lane>>4.
// VGPR j holds K = ((j<4)?0:16) + lh*8 + 2*(j&3), packed pair.
__device__ __forceinline__ void load_afrag(Frag16& f, const _Float16* rowbase, int lh) {
#pragma unroll
  for (int j = 0; j < 8; ++j) {
    int k = ((j < 4) ? 0 : 16) + lh * 8 + 2 * (j & 3);
    f.u[j] = *(const unsigned*)(rowbase + k);
  }
}

// B-fragment (32x16, KxN): lane lo = N column (caller passes that row of the
// K-contiguous storage), VGPR j holds K = ((j<4)?0:16) + lh*8 + 2*(j&3).
__device__ __forceinline__ void load_bfrag(Frag16& f, const _Float16* colbase, int lh) {
#pragma unroll
  for (int j = 0; j < 8; ++j) {
    int k = ((j < 4) ? 0 : 16) + lh * 8 + 2 * (j & 3);
    f.u[j] = *(const unsigned*)(colbase + k);
  }
}

// CDNA5 async global->LDS b128 (ASYNCcnt-tracked, no VGPR data staging).
// LDS offset = low 32 bits of the flat shared pointer (ISA 10.2 aperture rule).
__device__ __forceinline__ void async_b128(const _Float16* lds_dst, const _Float16* gsrc) {
  unsigned l = (unsigned)(size_t)lds_dst;
  asm volatile("global_load_async_to_lds_b128 %0, %1, off"
               :: "v"(l), "v"(gsrc) : "memory");
}

// ---------------------------------------------------------------------------
// fp32 -> f16 cast
// ---------------------------------------------------------------------------
__global__ void cast_f32_to_f16(const float* __restrict__ in,
                                _Float16* __restrict__ out, size_t n) {
  size_t i = (size_t)blockIdx.x * blockDim.x + threadIdx.x;
  size_t stride = (size_t)gridDim.x * blockDim.x;
  for (; i < n; i += stride) out[i] = (_Float16)in[i];
}

// ---------------------------------------------------------------------------
// C[M,N] = A[M,K] @ W[N,K]^T   (both K-contiguous), f16 in, f32 accum.
// Block: 256 threads = 8 waves; tile 128x128; wave tile 64(M) x 32(N).
// K-tile 64, LDS double-buffered, async-to-LDS staging.
// ---------------------------------------------------------------------------
#define TK  64
#define LDA 80   // halves; 160B row stride -> 16B aligned for b128

template <bool OUT_F32>
__global__ __launch_bounds__(256) void gemm_xwT(const _Float16* __restrict__ A,
                                                const _Float16* __restrict__ W,
                                                void* __restrict__ Cout,
                                                int M, int N, int K) {
  __shared__ __align__(16) _Float16 As[2][128][LDA];
  __shared__ __align__(16) _Float16 Bs[2][128][LDA];

  const int bm = blockIdx.y * 128;
  const int bn = blockIdx.x * 128;
  const int tid = threadIdx.x;
  const int wave = tid >> 5;
  const int lane = tid & 31;
  const int wm = wave >> 2;      // 0..1 -> 64 rows
  const int wn = wave & 3;       // 0..3 -> 32 cols
  const int lh = lane >> 4;
  const int lo = lane & 15;

  v8f acc[4][2] = {};

  // issue one tile: 1024 b128 vectors over 256 threads -> 8 asyncs/thread (4 A + 4 B)
  auto issue = [&](int buf, int k0) {
#pragma unroll
    for (int it = 0; it < 4; ++it) {
      int i = tid + it * 256;
      int r = i >> 3, c = (i & 7) * 8;
      async_b128(&As[buf][r][c], &A[(size_t)(bm + r) * K + k0 + c]);
      async_b128(&Bs[buf][r][c], &W[(size_t)(bn + r) * K + k0 + c]);
    }
  };

  issue(0, 0);
  int buf = 0;
  for (int k0 = 0; k0 < K; k0 += TK, buf ^= 1) {
    const bool has_next = (k0 + TK) < K;
    if (has_next) {
      issue(buf ^ 1, k0 + TK);
      // 16 outstanding -> wait until only next tile's 8 remain (in-order per wave)
      asm volatile("s_wait_asynccnt 0x8" ::: "memory");
    } else {
      asm volatile("s_wait_asynccnt 0x0" ::: "memory");
    }
    __syncthreads();   // current tile resident for all waves

#pragma unroll
    for (int ks = 0; ks < 2; ++ks) {
      Frag16 af[4], bf[2];
#pragma unroll
      for (int mt = 0; mt < 4; ++mt)
        load_afrag(af[mt], &As[buf][wm * 64 + mt * 16 + lo][ks * 32], lh);
#pragma unroll
      for (int nt = 0; nt < 2; ++nt)
        load_bfrag(bf[nt], &Bs[buf][wn * 32 + nt * 16 + lo][ks * 32], lh);
#pragma unroll
      for (int mt = 0; mt < 4; ++mt)
#pragma unroll
        for (int nt = 0; nt < 2; ++nt)
          acc[mt][nt] = wmma_f16(af[mt], bf[nt], acc[mt][nt]);
    }
    __syncthreads();   // all waves done reading buf before tile i+2 overwrites it
  }

#pragma unroll
  for (int mt = 0; mt < 4; ++mt)
#pragma unroll
    for (int nt = 0; nt < 2; ++nt)
#pragma unroll
      for (int r = 0; r < 8; ++r) {
        size_t m = bm + wm * 64 + mt * 16 + lh * 8 + r;
        size_t n = bn + wn * 32 + nt * 16 + lo;
        float v = acc[mt][nt][r];
        if (OUT_F32) ((float*)Cout)[m * N + n] = v;
        else         ((_Float16*)Cout)[m * N + n] = (_Float16)v;
      }
}

// ---------------------------------------------------------------------------
// RoPE + head permute: In [B][S][nh][64] -> Out [B][nh][S][64]
// ---------------------------------------------------------------------------
__global__ void rope_permute(const _Float16* __restrict__ in,
                             _Float16* __restrict__ out, int nh) {
  size_t idx = (size_t)blockIdx.x * blockDim.x + threadIdx.x;
  size_t total = (size_t)BB * SS * nh * 32;
  if (idx >= total) return;
  int j = idx & 31; size_t t = idx >> 5;
  int h = t % nh; t /= nh;
  int s = t % SS; int b = t / SS;
  const _Float16* row = in + (((size_t)b * SS + s) * nh + h) * HD;
  float x1 = (float)row[j], x2 = (float)row[j + 32];
  // inv_freq = 10000^(-2j/64)
  float th = (float)s * __expf(-(float)(2 * j) * (9.2103403719761836f / 64.0f));
  float c = __cosf(th), sn = __sinf(th);
  _Float16* orow = out + (((size_t)b * nh + h) * SS + s) * HD;
  orow[j]      = (_Float16)(x1 * c - x2 * sn);
  orow[j + 32] = (_Float16)(x2 * c + x1 * sn);
}

// V transpose: In [B][S][NKVH][64] -> Out [B][NKVH][64][S]
__global__ void v_transpose(const _Float16* __restrict__ in,
                            _Float16* __restrict__ out) {
  size_t idx = (size_t)blockIdx.x * blockDim.x + threadIdx.x;
  size_t total = (size_t)BB * SS * NKVH * HD;
  if (idx >= total) return;
  int d = idx & 63; size_t t = idx >> 6;
  int h = t % NKVH; t /= NKVH;
  int s = t % SS; int b = t / SS;
  out[(((size_t)b * NKVH + h) * HD + d) * SS + s] =
      in[(((size_t)b * SS + s) * NKVH + h) * HD + d];
}

// ---------------------------------------------------------------------------
// Causal flash attention, GQA.  Qh [B][NH][S][64], Kh [B][NKVH][S][64],
// Vt [B][NKVH][64][S], Attn out [B][S][NH*64] (f16).
// Block: 128 threads = 4 waves; 64 q-rows per block (16 per wave).
// ---------------------------------------------------------------------------
#define LDK 80   // halves; 160B stride -> 16B-aligned b128 LDS ops

__global__ __launch_bounds__(128) void flash_attn(const _Float16* __restrict__ Qh,
                                                  const _Float16* __restrict__ Kh,
                                                  const _Float16* __restrict__ Vt,
                                                  _Float16* __restrict__ Attn) {
  __shared__ __align__(16) _Float16 Ks[64][LDK];      // [key][d]
  __shared__ __align__(16) _Float16 Vs[64][LDK];      // [d][key]
  __shared__ __align__(16) _Float16 Ps[4][16][LDK];   // per-wave P tile [m][key]

  const int qt = blockIdx.x;        // q tile (64 rows)
  const int h  = blockIdx.y;        // head
  const int b  = blockIdx.z;
  const int kvh = h >> 2;           // 32 heads / 8 kv heads
  const int tid = threadIdx.x;
  const int wave = tid >> 5;
  const int lane = tid & 31;
  const int lh = lane >> 4;
  const int lo = lane & 15;

  // Q fragments for this wave's 16 rows (2 K-steps over d=64), straight from global.
  const _Float16* qbase =
      Qh + (((size_t)b * NH + h) * SS + qt * 64 + wave * 16 + lo) * HD;
  Frag16 qf[2];
#pragma unroll
  for (int ks = 0; ks < 2; ++ks) load_afrag(qf[ks], qbase + ks * 32, lh);

  const _Float16* kbase = Kh + ((size_t)b * NKVH + kvh) * SS * HD;
  const _Float16* vbase = Vt + ((size_t)b * NKVH + kvh) * HD * SS;

  float mrow[8], lrow[8];
  v8f o[4] = {};
#pragma unroll
  for (int r = 0; r < 8; ++r) { mrow[r] = -1e30f; lrow[r] = 0.0f; }

  const int qrow0 = qt * 64 + wave * 16 + lh * 8;   // + r = absolute q index

  for (int kt = 0; kt <= qt; ++kt) {
    __syncthreads();   // all waves done reading previous K/V tile
    // 512 vectors over 128 threads -> 4 iters, async K + async V each
#pragma unroll
    for (int it = 0; it < 4; ++it) {
      int i = tid + it * 128;
      int r = i >> 3, c = (i & 7) * 8;
      async_b128(&Ks[r][c], &kbase[(size_t)(kt * 64 + r) * HD + c]);
      async_b128(&Vs[r][c], &vbase[(size_t)r * SS + kt * 64 + c]);
    }
    asm volatile("s_wait_asynccnt 0x0" ::: "memory");
    __syncthreads();

    // S = Q K^T  (16 x 64 per wave)
    v8f st[4] = {};
#pragma unroll
    for (int ks = 0; ks < 2; ++ks)
#pragma unroll
      for (int nt = 0; nt < 4; ++nt) {
        Frag16 bk;
        load_bfrag(bk, &Ks[nt * 16 + lo][ks * 32], lh);
        st[nt] = wmma_f16(qf[ks], bk, st[nt]);
      }

    // scale + causal mask + tile row max
    float tmax[8];
#pragma unroll
    for (int r = 0; r < 8; ++r) tmax[r] = -1e30f;
#pragma unroll
    for (int nt = 0; nt < 4; ++nt) {
      int key = kt * 64 + nt * 16 + lo;
#pragma unroll
      for (int r = 0; r < 8; ++r) {
        float s = st[nt][r] * 0.125f;
        if (key > qrow0 + r) s = -1e30f;
        st[nt][r] = s;
        tmax[r] = fmaxf(tmax[r], s);
      }
    }
#pragma unroll
    for (int r = 0; r < 8; ++r) {
      float v = tmax[r];
      v = fmaxf(v, __shfl_xor(v, 1, 32));
      v = fmaxf(v, __shfl_xor(v, 2, 32));
      v = fmaxf(v, __shfl_xor(v, 4, 32));
      v = fmaxf(v, __shfl_xor(v, 8, 32));
      tmax[r] = v;
    }

    float alpha[8], rsum[8];
#pragma unroll
    for (int r = 0; r < 8; ++r) {
      float mnew = fmaxf(mrow[r], tmax[r]);
      alpha[r] = __expf(mrow[r] - mnew);
      mrow[r] = mnew;
      rsum[r] = 0.0f;
    }

    // P = exp(S - m), stash to LDS in row-major for A-fragment reload
#pragma unroll
    for (int nt = 0; nt < 4; ++nt)
#pragma unroll
      for (int r = 0; r < 8; ++r) {
        float p = __expf(st[nt][r] - mrow[r]);
        rsum[r] += p;
        Ps[wave][lh * 8 + r][nt * 16 + lo] = (_Float16)p;
      }
#pragma unroll
    for (int r = 0; r < 8; ++r) {
      float v = rsum[r];
      v += __shfl_xor(v, 1, 32);
      v += __shfl_xor(v, 2, 32);
      v += __shfl_xor(v, 4, 32);
      v += __shfl_xor(v, 8, 32);
      lrow[r] = lrow[r] * alpha[r] + v;
    }
#pragma unroll
    for (int nt = 0; nt < 4; ++nt)
#pragma unroll
      for (int r = 0; r < 8; ++r) o[nt][r] *= alpha[r];

    // wave-local LDS RAW fence (split dependency counter, CDNA5)
    asm volatile("s_wait_dscnt 0" ::: "memory");

    // O += P @ V
#pragma unroll
    for (int ks = 0; ks < 2; ++ks) {
      Frag16 pf;
      load_afrag(pf, &Ps[wave][lo][ks * 32], lh);
#pragma unroll
      for (int nt = 0; nt < 4; ++nt) {
        Frag16 bv;
        load_bfrag(bv, &Vs[nt * 16 + lo][ks * 32], lh);
        o[nt] = wmma_f16(pf, bv, o[nt]);
      }
    }
  }

  _Float16* obase = Attn + ((size_t)b * SS + qt * 64 + wave * 16) * HID + h * HD;
#pragma unroll
  for (int nt = 0; nt < 4; ++nt)
#pragma unroll
    for (int r = 0; r < 8; ++r) {
      float v = o[nt][r] / lrow[r];
      obase[(size_t)(lh * 8 + r) * HID + nt * 16 + lo] = (_Float16)v;
    }
}

// ---------------------------------------------------------------------------
extern "C" void kernel_launch(void* const* d_in, const int* in_sizes, int n_in,
                              void* d_out, int out_size, void* d_ws, size_t ws_size,
                              hipStream_t stream) {
  const float* hs = (const float*)d_in[0];
  // d_in[1] = attention_mask: exactly causal -> handled analytically
  const float* Wq = (const float*)d_in[2];
  const float* Wk = (const float*)d_in[3];
  const float* Wv = (const float*)d_in[4];
  const float* Wo = (const float*)d_in[5];
  float* out = (float*)d_out;

  const size_t M = (size_t)BB * SS;           // 4096
  char* w = (char*)d_ws;
  size_t off = 0;
  auto alloc = [&](size_t bytes) {
    void* p = w + off;
    off = (off + bytes + 255) & ~(size_t)255;
    return p;
  };
  _Float16* Xh  = (_Float16*)alloc(M * HID * 2);
  _Float16* Wqh = (_Float16*)alloc((size_t)HID * HID * 2);
  _Float16* Wkh = (_Float16*)alloc((size_t)KVDIM * HID * 2);
  _Float16* Wvh = (_Float16*)alloc((size_t)KVDIM * HID * 2);
  _Float16* Woh = (_Float16*)alloc((size_t)HID * HID * 2);
  _Float16* Qp  = (_Float16*)alloc(M * HID * 2);
  _Float16* Kp  = (_Float16*)alloc(M * KVDIM * 2);
  _Float16* Vp  = (_Float16*)alloc(M * KVDIM * 2);
  _Float16* Qh  = (_Float16*)alloc(M * HID * 2);
  _Float16* Kh  = (_Float16*)alloc(M * KVDIM * 2);
  _Float16* Vt  = (_Float16*)alloc(M * KVDIM * 2);
  _Float16* At  = (_Float16*)alloc(M * HID * 2);

  // 1) casts
  cast_f32_to_f16<<<2048, 256, 0, stream>>>(hs, Xh, M * HID);
  cast_f32_to_f16<<<2048, 256, 0, stream>>>(Wq, Wqh, (size_t)HID * HID);
  cast_f32_to_f16<<<1024, 256, 0, stream>>>(Wk, Wkh, (size_t)KVDIM * HID);
  cast_f32_to_f16<<<1024, 256, 0, stream>>>(Wv, Wvh, (size_t)KVDIM * HID);
  cast_f32_to_f16<<<2048, 256, 0, stream>>>(Wo, Woh, (size_t)HID * HID);

  // 2) projections
  gemm_xwT<false><<<dim3(HID / 128, M / 128), 256, 0, stream>>>(Xh, Wqh, Qp,
                                                                (int)M, HID, HID);
  gemm_xwT<false><<<dim3(KVDIM / 128, M / 128), 256, 0, stream>>>(Xh, Wkh, Kp,
                                                                  (int)M, KVDIM, HID);
  gemm_xwT<false><<<dim3(KVDIM / 128, M / 128), 256, 0, stream>>>(Xh, Wvh, Vp,
                                                                  (int)M, KVDIM, HID);

  // 3) RoPE + layout
  {
    size_t tq = (size_t)BB * SS * NH * 32;
    rope_permute<<<(tq + 255) / 256, 256, 0, stream>>>(Qp, Qh, NH);
    size_t tk = (size_t)BB * SS * NKVH * 32;
    rope_permute<<<(tk + 255) / 256, 256, 0, stream>>>(Kp, Kh, NKVH);
    size_t tv = (size_t)BB * SS * NKVH * HD;
    v_transpose<<<(tv + 255) / 256, 256, 0, stream>>>(Vp, Vt);
  }

  // 4) attention
  flash_attn<<<dim3(SS / 64, NH, BB), 128, 0, stream>>>(Qh, Kh, Vt, At);

  // 5) output projection -> fp32 d_out
  gemm_xwT<true><<<dim3(HID / 128, M / 128), 256, 0, stream>>>(At, Woh, out,
                                                               (int)M, HID, HID);
}
